// Net_34763465294339
// MI455X (gfx1250) — matrix operand order repaired and checked by reading, not compiled, not run on hardware
//
#include <hip/hip_runtime.h>
#include <hip/hip_bf16.h>

// CDNA5 / gfx1250, wave32. bf16 WMMA implicit-GEMM conv + energy reduction.

typedef __attribute__((ext_vector_type(16))) __bf16 v16bf;
typedef __attribute__((ext_vector_type(8)))  __bf16 v8bf;
typedef __attribute__((ext_vector_type(8)))  float  v8f;

// Pack two f32 -> one dword of two bf16 (v_cvt_pk_bf16_f32).
__device__ __forceinline__ unsigned int pk_bf16(float a, float b) {
  __hip_bfloat162 t = __float22bfloat162_rn(make_float2(a, b));
  unsigned int u;
  __builtin_memcpy(&u, &t, sizeof(u));
  return u;
}

// K-loop over all k*k taps for NQ concurrent output columns (one wave).
// A = 16 channels x 32 taps (from f32 weights, shifted by s=c%8 so that every
// B read is a 16B-aligned ds_read_b128). One A build feeds NQ WMMAs.
template<int K, int KSTEPS, int NQ>
__device__ __forceinline__ void gemm_cols(
    const __bf16* __restrict__ seg, const float* __restrict__ wch,
    int cb0, int li, int hi, int s, float rmask, float* __restrict__ sloc)
{
  v8f acc[NQ];
#pragma unroll
  for (int q = 0; q < NQ; ++q) {
    v8f z = {0.f,0.f,0.f,0.f,0.f,0.f,0.f,0.f};
    acc[q] = z;
  }

  for (int r = 0; r < K; ++r) {
    const float* wrow = wch + r * K;
    const __bf16* brow = seg + (li + r) * 256 + cb0;
    __builtin_prefetch(wrow + K, 0, 1);          // global_prefetch_b8: next row

#pragma unroll
    for (int kc = 0; kc < KSTEPS; ++kc) {
      const int kcol = kc * 32;
      union { v16bf v; unsigned int u[8]; } A;   // u[d] = elements {2d, 2d+1}
      const int t0 = kcol + (hi << 3) - s;       // taps t0+e and t0+16+e

      if (kcol >= 8 && kcol + 31 < K) {
        // Interior chunk (compile-time known): no clamping. 4B-aligned b128
        // loads are fine in gfx1250 DWORD alignment mode.
        const float4* pa = (const float4*)__builtin_assume_aligned(wrow + t0, 16);
        const float4* pb = (const float4*)__builtin_assume_aligned(wrow + t0 + 16, 16);
        float4 a0 = pa[0], a1 = pa[1], b0 = pb[0], b1 = pb[1];
        A.u[0] = pk_bf16(a0.x, a0.y);  A.u[1] = pk_bf16(a0.z, a0.w);
        A.u[2] = pk_bf16(a1.x, a1.y);  A.u[3] = pk_bf16(a1.z, a1.w);
        A.u[4] = pk_bf16(b0.x, b0.y);  A.u[5] = pk_bf16(b0.z, b0.w);
        A.u[6] = pk_bf16(b1.x, b1.y);  A.u[7] = pk_bf16(b1.z, b1.w);
      } else {
        // Boundary chunk: clamped scalar loads + f32 selects, then pack.
        float v[16];
#pragma unroll
        for (int e = 0; e < 8; ++e) {
          int ta = t0 + e;
          int tb = ta + 16;
          int tac = min(max(ta, 0), K - 1);
          int tbc = min(max(tb, 0), K - 1);
          float va = wrow[tac];
          float vb = wrow[tbc];
          v[e]     = (ta == tac) ? va : 0.0f;    // v_cndmask_b32
          v[e + 8] = (tb == tbc) ? vb : 0.0f;
        }
#pragma unroll
        for (int d = 0; d < 8; ++d) A.u[d] = pk_bf16(v[2*d], v[2*d + 1]);
      }

      const __bf16* bb = brow + kcol + (hi << 3);
#pragma unroll
      for (int q = 0; q < NQ; ++q) {
        union { v16bf v; v8bf h[2]; } B;
        B.h[0] = *(const v8bf*)(bb + 8 * q);       // ds_read_b128
        B.h[1] = *(const v8bf*)(bb + 8 * q + 16);  // ds_read_b128 (shared w/ q+2)
        acc[q] = __builtin_amdgcn_wmma_f32_16x16x32_bf16(
            false, A.v, false, B.v, (short)0, acc[q], false, false);
      }
    }
  }

#pragma unroll
  for (int q = 0; q < NQ; ++q) {
#pragma unroll
    for (int j = 0; j < 8; ++j) {
      float y = acc[q][j];
      sloc[j] += rmask * y * y;
    }
  }
}

// One block = 256 threads = 8 waves: (batch n, 16 output rows, column range).
// LDS: K+15 input rows as bf16 (pitch 256, 64-elem zero tail).
template<int K, int KSTEPS>
__global__ __launch_bounds__(256) void conv_energy_wmma(
    const float* __restrict__ x,      // (16,1,256,256) f32
    const float* __restrict__ w,      // (16,1,K,K) f32
    float* __restrict__ partial,      // per-(block,wave) 16-channel partials
    int outHW, int CS, int colsPerSplit)
{
  extern __shared__ __bf16 seg[];

  const int tid = threadIdx.x;
  const int gid = blockIdx.x;
  const int rowgroups = (outHW + 15) >> 4;
  const int perN  = rowgroups * CS;
  const int n     = gid / perN;
  const int rem   = gid - n * perN;
  const int rgI   = rem / CS;
  const int split = rem - rgI * CS;
  const int R0    = rgI * 16;
  const int total = (K + 15) * 256;
  const float* xb = x + n * 65536;

  // ---- Stage x rows [R0, R0+K+15) into LDS as bf16 ----
  for (int idx = tid * 2; idx < total; idx += 512) {
    int r   = idx >> 8;
    int col = idx & 255;
    int row = R0 + r;
    float a0 = 0.0f, a1 = 0.0f;
    if (row < 256) {
      const float* p = xb + row * 256 + col;
      a0 = p[0]; a1 = p[1];
    }
    *(unsigned int*)(seg + idx) = pk_bf16(a0, a1);
  }
  {
    unsigned int* segu = (unsigned int*)seg;
    for (int i = tid; i < 32; i += 256) segu[(total >> 1) + i] = 0u;
  }
  __syncthreads();

  const int lane = tid & 31;
  const int wv   = tid >> 5;     // wave id 0..7 == column residue s
  const int li   = lane & 15;
  const int hi   = lane >> 4;
  const int colLo = split * colsPerSplit;
  const int ncols = colsPerSplit >> 3;
  const float rmask = ((R0 + li) < outHW) ? 1.0f : 0.0f;
  const float* wch  = w + li * (K * K);

  float sloc[8];
#pragma unroll
  for (int j = 0; j < 8; ++j) sloc[j] = 0.0f;

  // Column groups: Q8 (pressure-gated) -> Q4 -> Q2 -> Q1, all exact.
  int g = 0;
  if constexpr (KSTEPS <= 4) {
    for (; g + 8 <= ncols; g += 8)
      gemm_cols<K, KSTEPS, 8>(seg, wch, colLo + 8 * g, li, hi, wv, rmask, sloc);
  }
  for (; g + 4 <= ncols; g += 4)
    gemm_cols<K, KSTEPS, 4>(seg, wch, colLo + 8 * g, li, hi, wv, rmask, sloc);
  if (ncols - g >= 2) {
    gemm_cols<K, KSTEPS, 2>(seg, wch, colLo + 8 * g, li, hi, wv, rmask, sloc);
    g += 2;
  }
  if (ncols - g >= 1)
    gemm_cols<K, KSTEPS, 1>(seg, wch, colLo + 8 * g, li, hi, wv, rmask, sloc);

  // Reduce over the 16 output rows (16-lane groups), wave32 butterflies.
#pragma unroll
  for (int j = 0; j < 8; ++j) {
    float v = sloc[j];
    v += __shfl_xor(v, 1, 32);
    v += __shfl_xor(v, 2, 32);
    v += __shfl_xor(v, 4, 32);
    v += __shfl_xor(v, 8, 32);
    sloc[j] = v;
  }
  if (li == 0) {  // lane 0: channels 0..7; lane 16: channels 8..15
    float* pw = partial + ((size_t)gid * 8 + wv) * 16 + hi * 8;
#pragma unroll
    for (int j = 0; j < 8; ++j) pw[j] = sloc[j];  // deterministic: no atomics
  }
}

// Fixed-order sum of per-wave partials; mean + scale + channel-half fold.
__global__ void finalize_kernel(const float* __restrict__ P, float* __restrict__ out) {
  int idx = blockIdx.x * blockDim.x + threadIdx.x;
  if (idx >= 512) return;
  int n    = idx >> 5;
  int ch   = (idx >> 2) & 7;
  int conv = idx & 3;
  const int   blocksPerN[4] = {15, 26, 20, 16};
  const int   base[4]  = {0, 30720, 30720 + 53248, 30720 + 53248 + 40960};
  const float denom[4] = {232.f*232.f*1.f, 208.f*208.f*2.f,
                          160.f*160.f*4.f, 64.f*64.f*8.f};
  const int bpn = blocksPerN[conv];
  const float* p = P + base[conv] + (size_t)(n * bpn) * 128;
  float s = 0.0f;
  for (int wvb = 0; wvb < bpn * 8; ++wvb)
    s += p[wvb * 16 + ch] + p[wvb * 16 + ch + 8];
  out[idx] = s / denom[conv];
}

extern "C" void kernel_launch(void* const* d_in, const int* in_sizes, int n_in,
                              void* d_out, int out_size, void* d_ws, size_t ws_size,
                              hipStream_t stream) {
  (void)in_sizes; (void)n_in; (void)out_size; (void)ws_size;
  const float* x  = (const float*)d_in[0];
  const float* w0 = (const float*)d_in[1];
  const float* w1 = (const float*)d_in[2];
  const float* w2 = (const float*)d_in[3];
  const float* w3 = (const float*)d_in[4];
  float* P   = (float*)d_ws;    // 157,696 floats of per-wave partials
  float* out = (float*)d_out;

  // lds bytes = ((K+15)*256 + 64) * 2
  conv_energy_wmma<25, 1><<<dim3(240), dim3(256), (size_t)((25+15)*256+64)*2, stream>>>(
      x, w0, P + 0, 232, 1, 232);
  conv_energy_wmma<49, 2><<<dim3(416), dim3(256), (size_t)((49+15)*256+64)*2, stream>>>(
      x, w1, P + 30720, 208, 2, 104);
  conv_energy_wmma<97, 4><<<dim3(320), dim3(256), (size_t)((97+15)*256+64)*2, stream>>>(
      x, w2, P + 30720 + 53248, 160, 2, 80);
  conv_energy_wmma<193, 7><<<dim3(256), dim3(256), (size_t)((193+15)*256+64)*2, stream>>>(
      x, w3, P + 30720 + 53248 + 40960, 64, 4, 16);

  finalize_kernel<<<dim3(2), dim3(256), 0, stream>>>(P, out);
}
